// GinAR_72146860638783
// MI455X (gfx1250) — matrix operand order repaired and checked by reading, not compiled, not run on hardware
//
#include <hip/hip_runtime.h>

typedef unsigned short u16;
typedef __attribute__((ext_vector_type(8)))  __bf16 v8bf;
typedef __attribute__((ext_vector_type(16))) __bf16 v16bf;
typedef __attribute__((ext_vector_type(8)))  float  v8f;

#define N_NODES 4096
#define BB 16
#define LL 12
#define CC 3
#define EE 32
#define GG 16
#define COLS1 (BB * LL * CC)   /* 576  */
#define COLS2 (BB * LL * EE)   /* 6144 */

__device__ __forceinline__ u16 f2bf(float f) {
  union { float f; unsigned u; } x; x.f = f;
  unsigned u = x.u;
  u += 0x7FFFu + ((u >> 16) & 1u);   // round-to-nearest-even
  return (u16)(u >> 16);
}

// Branch-free activations on the native v_exp2 / v_rcp transcendental units.
__device__ __forceinline__ float fast_exp(float x) {
#if __has_builtin(__builtin_amdgcn_exp2f)
  return __builtin_amdgcn_exp2f(x * 1.44269504088896340736f);
#else
  return __expf(x);
#endif
}
__device__ __forceinline__ float fast_rcp(float x) {
#if __has_builtin(__builtin_amdgcn_rcpf)
  return __builtin_amdgcn_rcpf(x);
#else
  return 1.0f / x;
#endif
}
__device__ __forceinline__ float sigm(float x) {
  return fast_rcp(1.0f + fast_exp(-x));
}
__device__ __forceinline__ float tanh_fast(float x) {
  float e = fast_exp(2.0f * x);           // saturates to +/-1 at |x| large
  return 1.0f - 2.0f * fast_rcp(e + 1.0f);
}

__device__ __forceinline__ v16bf cat16(v8bf a, v8bf b) {
  return __builtin_shufflevector(a, b, 0, 1, 2, 3, 4, 5, 6, 7, 8, 9, 10, 11, 12,
                                 13, 14, 15);
}

// ---------------------------------------------------------------------------
// A = adj + softmax(relu(e1 @ e2)) per row; also rowsum(A) for bias folding.
// ---------------------------------------------------------------------------
__global__ __launch_bounds__(256) void build_A_kernel(
    const float* __restrict__ adj, const float* __restrict__ e1,
    const float* __restrict__ e2, u16* __restrict__ Abf,
    float* __restrict__ rowsum) {
  const int n = blockIdx.x;
  const int tid = threadIdx.x;
  __shared__ float se1[GG];
  __shared__ float red[256];
  if (tid < GG) se1[tid] = e1[n * GG + tid];
  __syncthreads();

  float vals[16];
  float vmax = 0.0f;  // relu output >= 0
#pragma unroll
  for (int i = 0; i < 16; ++i) {
    int m = tid + i * 256;
    float s = 0.f;
#pragma unroll
    for (int g = 0; g < GG; ++g) s += se1[g] * e2[g * N_NODES + m];
    s = fmaxf(s, 0.f);
    vals[i] = s;
    vmax = fmaxf(vmax, s);
  }
  red[tid] = vmax; __syncthreads();
  for (int o = 128; o > 0; o >>= 1) {
    if (tid < o) red[tid] = fmaxf(red[tid], red[tid + o]);
    __syncthreads();
  }
  vmax = red[0]; __syncthreads();

  float esum = 0.f;
#pragma unroll
  for (int i = 0; i < 16; ++i) { vals[i] = fast_exp(vals[i] - vmax); esum += vals[i]; }
  red[tid] = esum; __syncthreads();
  for (int o = 128; o > 0; o >>= 1) {
    if (tid < o) red[tid] += red[tid + o];
    __syncthreads();
  }
  float inv = fast_rcp(red[0]);
  __syncthreads();

  float asum = 0.f;
#pragma unroll
  for (int i = 0; i < 16; ++i) {
    int m = tid + i * 256;
    float a = adj[(size_t)n * N_NODES + m];
    asum += a;
    Abf[(size_t)n * N_NODES + m] = f2bf(a + vals[i] * inv);
  }
  red[tid] = asum; __syncthreads();
  for (int o = 128; o > 0; o >>= 1) {
    if (tid < o) red[tid] += red[tid + o];
    __syncthreads();
  }
  if (tid == 0) rowsum[n] = red[0] + 1.0f;  // + softmax row sum (==1)
}

// ---------------------------------------------------------------------------
// Fold weights: W1 = W_emb1 @ W_g1 (f32), W2 col-major bf16, beW, Wd, bd.
// ---------------------------------------------------------------------------
__global__ __launch_bounds__(256) void precompute_kernel(
    const float* __restrict__ w_emb1, const float* __restrict__ b_emb1,
    const float* __restrict__ w_g1,
    const float* __restrict__ w_emb2, const float* __restrict__ b_emb2,
    const float* __restrict__ w_g2,
    const float* __restrict__ w_dec, const float* __restrict__ b_dec,
    const float* __restrict__ w_out, const float* __restrict__ b_out,
    float* __restrict__ W1, float* __restrict__ beW1,
    u16* __restrict__ W2cm, float* __restrict__ beW2,
    float* __restrict__ Wd, float* __restrict__ bd) {
  int tid = threadIdx.x;
  if (tid < 128) {
    int j = tid;
    for (int c = 0; c < CC; ++c) {
      float s = 0.f;
      for (int e = 0; e < EE; ++e) s += w_emb1[c * EE + e] * w_g1[e * 128 + j];
      W1[c * 128 + j] = s;
    }
    float s1 = 0.f;
    for (int e = 0; e < EE; ++e) s1 += b_emb1[e] * w_g1[e * 128 + j];
    beW1[j] = s1;
    for (int ep = 0; ep < EE; ++ep) {
      float s = 0.f;
      for (int e = 0; e < EE; ++e) s += w_emb2[ep * EE + e] * w_g2[e * 128 + j];
      W2cm[j * EE + ep] = f2bf(s);   // col-major bf16 for WMMA B-fragments
    }
    float s2 = 0.f;
    for (int e = 0; e < EE; ++e) s2 += b_emb2[e] * w_g2[e * 128 + j];
    beW2[j] = s2;
  }
  for (int i = tid; i < 12 * 64; i += 256) {
    int o2 = i >> 6, ze = i & 63, z = ze >> 5, e = ze & 31;
    float s = 0.f;
    for (int o = 0; o < 12; ++o) s += w_out[o2 * 12 + o] * w_dec[(o * 2 + z) * EE + e];
    Wd[i] = s;
  }
  if (tid < 12) {
    float s = b_out[tid];
    for (int o = 0; o < 12; ++o) s += w_out[tid * 12 + o] * b_dec[o];
    bd[tid] = s;
  }
}

// ---------------------------------------------------------------------------
// Pack history [B,L,N,C] -> X1 bf16 [N][B*L*C]
// ---------------------------------------------------------------------------
__global__ void pack_X1_kernel(const float* __restrict__ hist, u16* __restrict__ X1) {
  int id = blockIdx.x * 256 + threadIdx.x;
  if (id >= N_NODES * COLS1) return;
  int m = id / COLS1;
  int col = id - m * COLS1;
  int b = col / (LL * CC);
  int rem = col - b * (LL * CC);
  int l = rem / CC;
  int c = rem - l * CC;
  float v = hist[(((size_t)(b * LL + l)) * N_NODES + m) * CC + c];
  X1[(size_t)m * COLS1 + col] = f2bf(v);
}

// ---------------------------------------------------------------------------
// bf16 WMMA GEMM: Y[OutT, MMxNN] = A[bf16, MMxK] * X[bf16, KxNN]
// Block = 256 thr (8 waves); wave tile = 16 x BN; block tile = 128 x BN.
// Double-buffered LDS staging, pipelined A-frags, paired B-frag loads with a
// scheduling fence so the second WMMA of each pair issues wait-free.
// ---------------------------------------------------------------------------
template <int BN, typename OutT>
__global__ __launch_bounds__(256) void gemm_bf16_wmma(
    const u16* __restrict__ A, const u16* __restrict__ X,
    OutT* __restrict__ Y, int K, int NN) {
  constexpr int NFRAG  = BN / 16;    // 16x16 col tiles per wave
  constexpr int TPC    = BN / 2;     // threads per row-slab (2 cols each)
  constexpr int SLABS  = 256 / TPC;  // row-slabs in staging map
  constexpr int RPT    = 32 / SLABS; // rows per staging thread
  constexpr int STRIDE = 40;         // u16 per col: 32 K-elems + 8 pad
  __shared__ u16 ldsT[2][BN * STRIDE];

  const int tid  = threadIdx.x;
  const int wave = tid >> 5;
  const int lane = tid & 31;
  const int l16  = lane & 15;
  const int half = lane >> 4;
  const int rowBase  = blockIdx.x * 128 + wave * 16;
  const int colBlock = blockIdx.y * BN;

  const int c2 = tid % TPC;
  const int r0 = (tid / TPC) * RPT;

  v8f acc[NFRAG] = {};
  const size_t arow = (size_t)(rowBase + l16) * (size_t)K;

  unsigned raw[RPT];
  // ---- prologue: stage tile 0, load A-frag 0 ----
#pragma unroll
  for (int r = 0; r < RPT; ++r)
    raw[r] = *(const unsigned*)(X + (size_t)(r0 + r) * NN + colBlock + c2 * 2);
  {
    u16* d0 = &ldsT[0][(c2 * 2 + 0) * STRIDE + r0];
    u16* d1 = &ldsT[0][(c2 * 2 + 1) * STRIDE + r0];
#pragma unroll
    for (int w = 0; w < RPT / 2; ++w) {
      unsigned lo = (raw[2 * w] & 0xffffu) | (raw[2 * w + 1] << 16);
      unsigned hi = (raw[2 * w] >> 16) | (raw[2 * w + 1] & 0xffff0000u);
      ((unsigned*)d0)[w] = lo;
      ((unsigned*)d1)[w] = hi;
    }
  }
  v16bf aCur = cat16(*(const v8bf*)(A + arow + half * 8),
                     *(const v8bf*)(A + arow + 16 + half * 8));
  __syncthreads();

  for (int it = 0;; ++it) {
    const int k0 = it * 32;
    const bool more = (k0 + 32) < K;
    v16bf aNext = {};
    if (more) {
#pragma unroll
      for (int r = 0; r < RPT; ++r)
        raw[r] = *(const unsigned*)(X + (size_t)(k0 + 32 + r0 + r) * NN +
                                    colBlock + c2 * 2);
      aNext = cat16(*(const v8bf*)(A + arow + k0 + 32 + half * 8),
                    *(const v8bf*)(A + arow + k0 + 48 + half * 8));
      if (k0 + 96 < K) __builtin_prefetch(A + arow + k0 + 96, 0, 3);
    }

    const u16* bufc = ldsT[it & 1];
#pragma unroll
    for (int c = 0; c < NFRAG; c += 2) {
      const u16* bpA = bufc + (size_t)((c + 0) * 16 + l16) * STRIDE + half * 8;
      const u16* bpB = bufc + (size_t)((c + 1) * 16 + l16) * STRIDE + half * 8;
      v16bf bfA = cat16(((const v8bf*)bpA)[0], ((const v8bf*)bpA)[2]);
      v16bf bfB = cat16(((const v8bf*)bpB)[0], ((const v8bf*)bpB)[2]);
#if __has_builtin(__builtin_amdgcn_sched_barrier)
      __builtin_amdgcn_sched_barrier(0);  // keep both frags live before WMMA pair
#endif
      acc[c + 0] = __builtin_amdgcn_wmma_f32_16x16x32_bf16(
          false, aCur, false, bfA, (short)0, acc[c + 0], false, false);
      acc[c + 1] = __builtin_amdgcn_wmma_f32_16x16x32_bf16(
          false, aCur, false, bfB, (short)0, acc[c + 1], false, false);
    }

    if (!more) break;

    {
      u16* d0 = &ldsT[(it + 1) & 1][(c2 * 2 + 0) * STRIDE + r0];
      u16* d1 = &ldsT[(it + 1) & 1][(c2 * 2 + 1) * STRIDE + r0];
#pragma unroll
      for (int w = 0; w < RPT / 2; ++w) {
        unsigned lo = (raw[2 * w] & 0xffffu) | (raw[2 * w + 1] << 16);
        unsigned hi = (raw[2 * w] >> 16) | (raw[2 * w + 1] & 0xffff0000u);
        ((unsigned*)d0)[w] = lo;
        ((unsigned*)d1)[w] = hi;
      }
    }
    __syncthreads();
    aCur = aNext;
  }

  // ---- epilogue: C/D layout -> VGPR r = M row, lane = N col ----
#pragma unroll
  for (int c = 0; c < NFRAG; ++c) {
#pragma unroll
    for (int r = 0; r < 8; ++r) {
      int row = rowBase + r + half * 8;
      float v = acc[c][r];
      if constexpr (sizeof(OutT) == 2)
        Y[(size_t)row * NN + colBlock + c * 16 + l16] = f2bf(v);
      else
        Y[(size_t)row * NN + colBlock + c * 16 + l16] = v;
    }
  }
}

// ---------------------------------------------------------------------------
// LSTM layer 1 recurrence: gates from Y1 (3 channels) via folded W1.
// ---------------------------------------------------------------------------
__global__ __launch_bounds__(256) void lstm1_kernel(
    const float* __restrict__ Y1, const float* __restrict__ rowsum,
    const float* __restrict__ W1, const float* __restrict__ beW1,
    const float* __restrict__ b_g, u16* __restrict__ X2,
    float* __restrict__ last1) {
  __shared__ float sW[CC * 128];
  __shared__ float sbias[128];
  __shared__ float sbe[128];
  int tid = threadIdx.x;
  for (int i = tid; i < CC * 128; i += 256) sW[i] = W1[i];
  if (tid < 128) { sbias[tid] = b_g[tid]; sbe[tid] = beW1[tid]; }
  __syncthreads();

  int id = blockIdx.x * 256 + tid;
  int n = id & (N_NODES - 1);
  int b = id >> 12;
  float rs = rowsum[n];
  float cst[EE];
#pragma unroll
  for (int e = 0; e < EE; ++e) cst[e] = 0.f;

  const float* yrow = Y1 + (size_t)n * COLS1 + b * (LL * CC);
  u16* xout = X2 + (size_t)n * COLS2 + b * (LL * EE);
  for (int t = 0; t < LL; ++t) {
    float y0 = yrow[t * 3 + 0], y1 = yrow[t * 3 + 1], y2 = yrow[t * 3 + 2];
#pragma unroll
    for (int e = 0; e < EE; ++e) {
      float gf = sbias[e]      + rs * sbe[e]      + y0 * sW[e]      + y1 * sW[128 + e]      + y2 * sW[256 + e];
      float gi = sbias[e + 32] + rs * sbe[e + 32] + y0 * sW[e + 32] + y1 * sW[128 + e + 32] + y2 * sW[256 + e + 32];
      float go = sbias[e + 64] + rs * sbe[e + 64] + y0 * sW[e + 64] + y1 * sW[128 + e + 64] + y2 * sW[256 + e + 64];
      float gc = sbias[e + 96] + rs * sbe[e + 96] + y0 * sW[e + 96] + y1 * sW[128 + e + 96] + y2 * sW[256 + e + 96];
      cst[e] = sigm(gf) * cst[e] + sigm(gi) * tanh_fast(gc);
      float h = sigm(go) * tanh_fast(cst[e]);
      xout[t * EE + e] = f2bf(h);
      if (t == LL - 1) last1[(size_t)id * EE + e] = h;
    }
  }
}

// ---------------------------------------------------------------------------
// LSTM layer 2 on the matrix pipe: per timestep, gate tile (16 seqs x 128) =
// one 16x32 A-frag from bf16 Y2 times 8 B-frags of col-major bf16 W2.
// WMMA C/D layout (lane = gate col, VGPR row = sequence) feeds the
// element-wise LSTM update directly in registers.
// ---------------------------------------------------------------------------
__global__ __launch_bounds__(256) void lstm2_wmma_kernel(
    const u16* __restrict__ Y2bf, const float* __restrict__ rowsum,
    const u16* __restrict__ W2cm, const float* __restrict__ beW2,
    const float* __restrict__ b_g, float* __restrict__ last2) {
  __shared__ u16 sW[128 * EE];  // 8KB, col-major [j][k]
  int tid = threadIdx.x;
  for (int i = tid; i < 128 * EE / 2; i += 256)
    ((unsigned*)sW)[i] = ((const unsigned*)W2cm)[i];
  __syncthreads();

  const int wave = tid >> 5;
  const int lane = tid & 31;
  const int l16  = lane & 15;
  const int half = lane >> 4;
  const int id0  = (blockIdx.x * 8 + wave) * 16;  // 16 sequences per wave
  const int b    = id0 >> 12;
  const int n0   = id0 & (N_NODES - 1);

  float sb[8], se[8];
#pragma unroll
  for (int tl = 0; tl < 8; ++tl) {
    int j = tl * 16 + l16;
    sb[tl] = b_g[j];
    se[tl] = beW2[j];
  }
  float rsv[8];
#pragma unroll
  for (int r = 0; r < 8; ++r) rsv[r] = rowsum[n0 + r + half * 8];

  const size_t ybase = (size_t)(n0 + l16) * COLS2 + (size_t)b * (LL * EE);

  float c0[8], c1[8], h0[8], h1[8];
#pragma unroll
  for (int r = 0; r < 8; ++r) { c0[r] = 0.f; c1[r] = 0.f; h0[r] = 0.f; h1[r] = 0.f; }

  for (int t = 0; t < LL; ++t) {
    v16bf af = cat16(*(const v8bf*)(Y2bf + ybase + t * EE + half * 8),
                     *(const v8bf*)(Y2bf + ybase + t * EE + 16 + half * 8));
    v8f g[8];
#pragma unroll
    for (int tl = 0; tl < 8; ++tl) {
      const u16* bp = sW + (size_t)(tl * 16 + l16) * EE + half * 8;
      v16bf bfm = cat16(((const v8bf*)bp)[0], ((const v8bf*)bp)[2]);
      v8f z = {};
      g[tl] = __builtin_amdgcn_wmma_f32_16x16x32_bf16(
          false, af, false, bfm, (short)0, z, false, false);
    }
    // gate tiles: f -> 0,1  i -> 2,3  o -> 4,5  c -> 6,7
#pragma unroll
    for (int r = 0; r < 8; ++r) {
      float f0 = sigm(g[0][r] + sb[0] + rsv[r] * se[0]);
      float i0 = sigm(g[2][r] + sb[2] + rsv[r] * se[2]);
      float q0 = tanh_fast(g[6][r] + sb[6] + rsv[r] * se[6]);
      c0[r] = f0 * c0[r] + i0 * q0;
      float f1 = sigm(g[1][r] + sb[1] + rsv[r] * se[1]);
      float i1 = sigm(g[3][r] + sb[3] + rsv[r] * se[3]);
      float q1 = tanh_fast(g[7][r] + sb[7] + rsv[r] * se[7]);
      c1[r] = f1 * c1[r] + i1 * q1;
      if (t == LL - 1) {  // layer-2 h only consumed at the last timestep
        h0[r] = sigm(g[4][r] + sb[4] + rsv[r] * se[4]) * tanh_fast(c0[r]);
        h1[r] = sigm(g[5][r] + sb[5] + rsv[r] * se[5]) * tanh_fast(c1[r]);
      }
    }
  }
#pragma unroll
  for (int r = 0; r < 8; ++r) {
    int seq = id0 + r + half * 8;
    last2[(size_t)seq * EE + l16]      = h0[r];
    last2[(size_t)seq * EE + 16 + l16] = h1[r];
  }
}

// ---------------------------------------------------------------------------
// Fused decode: out[b,o,n] = sum_{z,e} last_z[b,n,e] * Wd[o,z,e] + bd[o]
// ---------------------------------------------------------------------------
__global__ __launch_bounds__(256) void decode_kernel(
    const float* __restrict__ last1, const float* __restrict__ last2,
    const float* __restrict__ Wd, const float* __restrict__ bd,
    float* __restrict__ out) {
  __shared__ float sWd[12 * 64];
  __shared__ float sbd[12];
  int tid = threadIdx.x;
  for (int i = tid; i < 12 * 64; i += 256) sWd[i] = Wd[i];
  if (tid < 12) sbd[tid] = bd[tid];
  __syncthreads();

  int id = blockIdx.x * 256 + tid;
  int n = id & (N_NODES - 1);
  int b = id >> 12;
  float f1[EE], f2[EE];
#pragma unroll
  for (int e = 0; e < EE; ++e) {
    f1[e] = last1[(size_t)id * EE + e];
    f2[e] = last2[(size_t)id * EE + e];
  }
  for (int o = 0; o < 12; ++o) {
    float s = sbd[o];
#pragma unroll
    for (int e = 0; e < EE; ++e)
      s += f1[e] * sWd[o * 64 + e] + f2[e] * sWd[o * 64 + 32 + e];
    out[((size_t)b * 12 + o) * N_NODES + n] = s;
  }
}

// ---------------------------------------------------------------------------
extern "C" void kernel_launch(void* const* d_in, const int* in_sizes, int n_in,
                              void* d_out, int out_size, void* d_ws, size_t ws_size,
                              hipStream_t stream) {
  (void)in_sizes; (void)n_in; (void)out_size; (void)ws_size;
  const float* history = (const float*)d_in[0];
  const float* adj     = (const float*)d_in[1];
  const float* w_emb1  = (const float*)d_in[2];
  const float* b_emb1  = (const float*)d_in[3];
  const float* e1_1    = (const float*)d_in[4];
  const float* e2_1    = (const float*)d_in[5];
  const float* w_g1    = (const float*)d_in[6];
  const float* b_g1    = (const float*)d_in[7];
  const float* w_emb2  = (const float*)d_in[8];
  const float* b_emb2  = (const float*)d_in[9];
  const float* e1_2    = (const float*)d_in[10];
  const float* e2_2    = (const float*)d_in[11];
  const float* w_g2    = (const float*)d_in[12];
  const float* b_g2    = (const float*)d_in[13];
  const float* w_dec   = (const float*)d_in[14];
  const float* b_dec   = (const float*)d_in[15];
  const float* w_out   = (const float*)d_in[16];
  const float* b_out   = (const float*)d_in[17];
  float* out = (float*)d_out;

  char* base = (char*)d_ws;
  size_t off = 0;
  auto take = [&](size_t bytes) -> void* {
    void* r = base + off;
    off += (bytes + 255) & ~(size_t)255;
    return r;
  };
  u16*   Abf   = (u16*)  take((size_t)N_NODES * N_NODES * 2);  // A1 then A2
  float* rs1   = (float*)take(N_NODES * 4);
  float* rs2   = (float*)take(N_NODES * 4);
  u16*   X1    = (u16*)  take((size_t)N_NODES * COLS1 * 2);
  float* Y1    = (float*)take((size_t)N_NODES * COLS1 * 4);
  u16*   X2    = (u16*)  take((size_t)N_NODES * COLS2 * 2);
  u16*   Y2    = (u16*)  take((size_t)N_NODES * COLS2 * 2);   // bf16
  float* W1    = (float*)take(CC * 128 * 4);
  float* beW1  = (float*)take(128 * 4);
  u16*   W2cm  = (u16*)  take(128 * EE * 2);
  float* beW2  = (float*)take(128 * 4);
  float* Wd    = (float*)take(12 * 64 * 4);
  float* bd    = (float*)take(12 * 4);
  float* last1 = (float*)take((size_t)BB * N_NODES * EE * 4);
  float* last2 = (float*)take((size_t)BB * N_NODES * EE * 4);

  precompute_kernel<<<1, 256, 0, stream>>>(w_emb1, b_emb1, w_g1, w_emb2, b_emb2,
                                           w_g2, w_dec, b_dec, w_out, b_out,
                                           W1, beW1, W2cm, beW2, Wd, bd);
  build_A_kernel<<<N_NODES, 256, 0, stream>>>(adj, e1_1, e2_1, Abf, rs1);
  pack_X1_kernel<<<(N_NODES * COLS1 + 255) / 256, 256, 0, stream>>>(history, X1);
  gemm_bf16_wmma<64, float><<<dim3(N_NODES / 128, COLS1 / 64), 256, 0, stream>>>(
      Abf, X1, Y1, N_NODES, COLS1);
  lstm1_kernel<<<BB * N_NODES / 256, 256, 0, stream>>>(Y1, rs1, W1, beW1, b_g1,
                                                       X2, last1);
  build_A_kernel<<<N_NODES, 256, 0, stream>>>(adj, e1_2, e2_2, Abf, rs2);
  gemm_bf16_wmma<128, u16><<<dim3(N_NODES / 128, COLS2 / 128), 256, 0, stream>>>(
      Abf, X2, Y2, N_NODES, COLS2);
  lstm2_wmma_kernel<<<BB * N_NODES / (16 * 8), 256, 0, stream>>>(
      Y2, rs2, W2cm, beW2, b_g2, last2);
  decode_kernel<<<BB * N_NODES / 256, 256, 0, stream>>>(last1, last2, Wd, bd, out);
}
